// BioRNN_working_42142219109075
// MI455X (gfx1250) — matrix operand order repaired and checked by reading, not compiled, not run on hardware
//
#include <hip/hip_runtime.h>
#include <hip/hip_bf16.h>

// ---- architecture constants (match reference) ----
#define TT     200
#define BB     128
#define NIN    64
#define NOUT   10
#define NES    512
#define NN     3840
#define SR_ED0   512
#define SR_ED1   1536
#define PFC_ES0  1920
#define PFC_ED0  2432
#define PFC_ED1  3456
#define DECAYF   0.2f
#define NOISE_SCALE 0.00632455532f   // sqrt(2*0.2)*0.01

// ---- GEMM tiling ----
#define WPT     3                 // 16x16 n-tiles per wave (16x48 per wave)
#define KCHUNK  128               // K elements staged in LDS per stage
#define NSTAGE  (NN / KCHUNK)     // 30
#define KSTEPS  (NN / 32)         // 120 wmma k-steps
#define PADK    (KCHUNK + 8)      // LDS row pitch (272 B): conflict-free ds_load_b128

typedef __attribute__((ext_vector_type(16))) __bf16 bf16x16;
typedef __attribute__((ext_vector_type(8)))  __bf16 bf16x8;
typedef __attribute__((ext_vector_type(8)))  float  f32x8;
typedef __attribute__((ext_vector_type(4)))  int    v4i;

#if defined(__gfx1250__) && __has_builtin(__builtin_amdgcn_global_load_async_to_lds_b128)
#define HAVE_ASYNC_LDS 1
#endif

static __device__ __forceinline__ bool is_dend_idx(int n) {
    return (n >= SR_ED0 && n < SR_ED1) || (n >= PFC_ED0 && n < PFC_ED1);
}

// 16-byte global -> LDS copy (async on CDNA5, sync fallback elsewhere)
static __device__ __forceinline__ void copy16_g2lds(const __hip_bfloat16* gsrc,
                                                    __hip_bfloat16* ldst) {
#if defined(HAVE_ASYNC_LDS)
    __builtin_amdgcn_global_load_async_to_lds_b128(
        (__attribute__((address_space(1))) v4i*)gsrc,
        (__attribute__((address_space(3))) v4i*)ldst,
        0, 0);
#else
    *(uint4*)ldst = *(const uint4*)gsrc;
#endif
}

static __device__ __forceinline__ void wait_async_done() {
#if defined(HAVE_ASYNC_LDS)
 #if __has_builtin(__builtin_amdgcn_s_wait_asynccnt)
    __builtin_amdgcn_s_wait_asynccnt(0);
 #else
    asm volatile("s_wait_asynccnt 0x0" ::: "memory");
 #endif
#endif
}

static __device__ __forceinline__ bf16x16 load_frag16(const __hip_bfloat16* p) {
    bf16x8 lo = *(const bf16x8*)(p);
    bf16x8 hi = *(const bf16x8*)(p + 8);
    return __builtin_shufflevector(lo, hi, 0,1,2,3,4,5,6,7,8,9,10,11,12,13,14,15);
}

// ---------------------------------------------------------------------------
// Precompute: w_effT[n*NN + k] = bf16(|w_rec[k,n]| * mask[k,n])
// ---------------------------------------------------------------------------
__global__ void build_weffT(const float* __restrict__ w_rec,
                            const float* __restrict__ mask,
                            __hip_bfloat16* __restrict__ wT) {
    long idx = (long)blockIdx.x * blockDim.x + threadIdx.x;
    if (idx >= (long)NN * NN) return;
    int n = (int)(idx / NN);
    int k = (int)(idx % NN);
    float v = fabsf(w_rec[(long)k * NN + n]) * mask[(long)k * NN + n];
    wT[idx] = __float2bfloat16(v);
}

// ---------------------------------------------------------------------------
// Init state: h = h0; r0 = activate(h0); keep f32 + bf16 copies of r.
// ---------------------------------------------------------------------------
__global__ void init_state(const float* __restrict__ h0,
                           float* __restrict__ h,
                           float* __restrict__ r,
                           __hip_bfloat16* __restrict__ rbf) {
    int idx = blockIdx.x * blockDim.x + threadIdx.x;
    if (idx >= BB * NN) return;
    int n = idx % NN;
    float hv = h0[idx];
    h[idx] = hv;
    float rv = is_dend_idx(n) ? tanhf(hv) : fmaxf(hv, 0.0f);
    r[idx] = rv;
    rbf[idx] = __float2bfloat16(rv);
}

// ---------------------------------------------------------------------------
// WMMA GEMM: total[b, n] = sum_k rbf[b, k] * wT[n, k]
//
// Workgroup = 8 waves, one 16-row M tile (blockIdx.y). A tile (16 x KCHUNK)
// is staged into LDS with async global->LDS copies, double-buffered across
// stages (s_wait_asynccnt + barrier). Each wave owns 16x48 of output:
// 3 accumulators, 3 WMMAs per k-step, B fragments double-buffered in
// registers so loads for k+1 overlap the WMMAs of k.
// ---------------------------------------------------------------------------
__global__ __launch_bounds__(256)
void gemm_rw(const __hip_bfloat16* __restrict__ rbf,   // [BB][NN]
             const __hip_bfloat16* __restrict__ wT,    // [NN][NN] transposed
             float* __restrict__ total) {              // [BB][NN]
    __shared__ __hip_bfloat16 smem[2][16 * PADK];      // 2 x 4.25 KB

    const int tid  = threadIdx.x;
    const int wave = tid >> 5;
    const int lane = tid & 31;
    const int half = lane >> 4;
    const int l    = lane & 15;

    const int m0 = blockIdx.y * 16;
    const int n0 = (blockIdx.x * 8 + wave) * (WPT * 16);

    // --- async copy mapping: 256 threads x 16B = one 16xKCHUNK stage ---
    const int crow = tid >> 4;     // 0..15 : A row within tile
    const int cseg = tid & 15;     // 0..15 : 8-element segment within KCHUNK
    const __hip_bfloat16* gsrc_base = rbf + (long)(m0 + crow) * NN + cseg * 8;
    __hip_bfloat16* ldst_base[2] = {
        &smem[0][crow * PADK + cseg * 8],
        &smem[1][crow * PADK + cseg * 8]
    };

    // --- B base pointers (lane-resident rows of wT) ---
    const __hip_bfloat16* bptr[WPT];
    #pragma unroll
    for (int nt = 0; nt < WPT; ++nt)
        bptr[nt] = wT + (long)(n0 + nt * 16 + l) * NN + half * 16;

    f32x8 acc[WPT];
    #pragma unroll
    for (int i = 0; i < WPT; ++i)
        acc[i] = (f32x8){0.f,0.f,0.f,0.f,0.f,0.f,0.f,0.f};

    // prologue: stage 0 A tile, k-step 0 B fragments
    copy16_g2lds(gsrc_base, ldst_base[0]);
    bf16x16 bcur[WPT], bnxt[WPT];
    #pragma unroll
    for (int nt = 0; nt < WPT; ++nt) bcur[nt] = load_frag16(bptr[nt]);

    int kstep = 0;
    for (int s = 0; s < NSTAGE; ++s) {
        wait_async_done();        // my portion of stage s arrived
        __syncthreads();          // all portions arrived; prev buffer free
        if (s + 1 < NSTAGE)       // overlap next stage copy with compute
            copy16_g2lds(gsrc_base + (long)(s + 1) * KCHUNK,
                         ldst_base[(s + 1) & 1]);

        const __hip_bfloat16* abuf = &smem[s & 1][0];
        #pragma unroll
        for (int kk = 0; kk < KCHUNK / 32; ++kk) {
            // issue B loads for the next k-step before computing this one
            if (kstep + 1 < KSTEPS) {
                const long koff = (long)(kstep + 1) * 32;
                #pragma unroll
                for (int nt = 0; nt < WPT; ++nt)
                    bnxt[nt] = load_frag16(bptr[nt] + koff);
            }
            // A fragment from LDS (conflict-free: row pitch 272 B)
            const __hip_bfloat16* ap = abuf + l * PADK + kk * 32 + half * 16;
            bf16x8 alo = *(const bf16x8*)(ap);
            bf16x8 ahi = *(const bf16x8*)(ap + 8);
            bf16x16 a = __builtin_shufflevector(alo, ahi,
                          0,1,2,3,4,5,6,7,8,9,10,11,12,13,14,15);
            #pragma unroll
            for (int nt = 0; nt < WPT; ++nt)
                acc[nt] = __builtin_amdgcn_wmma_f32_16x16x32_bf16(
                    false, a, false, bcur[nt], (short)0, acc[nt], false, false);
            #pragma unroll
            for (int nt = 0; nt < WPT; ++nt) bcur[nt] = bnxt[nt];
            ++kstep;
        }
    }

    // C/D layout: VGPR j -> row m0 + half*8 + j, col n0 + nt*16 + l
    #pragma unroll
    for (int nt = 0; nt < WPT; ++nt) {
        #pragma unroll
        for (int j = 0; j < 8; ++j) {
            int row = m0 + half * 8 + j;
            int col = n0 + nt * 16 + l;
            total[(long)row * NN + col] = acc[nt][j];
        }
    }
}

// ---------------------------------------------------------------------------
// Pointwise step: total += x_t@w_in + b + dend->soma; leaky-integrate + noise;
// activation; emit f32 + bf16 copies of new rate.
// ---------------------------------------------------------------------------
__global__ void step_update(const float* __restrict__ total,   // [BB][NN]
                            const float* __restrict__ x_t,     // [BB][NIN]
                            const float* __restrict__ noise_t, // [BB][NN]
                            const float* __restrict__ w_in,    // [NIN][NN]
                            const float* __restrict__ bias,    // [NN]
                            const float* __restrict__ r_old,   // [BB][NN]
                            float* __restrict__ h,             // [BB][NN] in/out
                            float* __restrict__ r_new,         // [BB][NN]
                            __hip_bfloat16* __restrict__ rbf_new) {
    int idx = blockIdx.x * blockDim.x + threadIdx.x;
    if (idx >= BB * NN) return;
    int b = idx / NN;
    int n = idx % NN;

    float tot = total[idx] + bias[n];

    const float* xb = x_t + b * NIN;
    #pragma unroll 8
    for (int i = 0; i < NIN; ++i)
        tot = fmaf(xb[i], w_in[(long)i * NN + n], tot);

    // additive dendrite -> soma coupling (N_BR = 2, branch-major)
    if (n < NES) {
        tot += r_old[(long)b * NN + SR_ED0 + n] +
               r_old[(long)b * NN + SR_ED0 + NES + n];
    } else if (n >= PFC_ES0 && n < PFC_ES0 + NES) {
        int e = n - PFC_ES0;
        tot += r_old[(long)b * NN + PFC_ED0 + e] +
               r_old[(long)b * NN + PFC_ED0 + NES + e];
    }

    float hn = (1.0f - DECAYF) * h[idx] + DECAYF * tot + NOISE_SCALE * noise_t[idx];
    h[idx] = hn;
    float rn = is_dend_idx(n) ? tanhf(hn) : fmaxf(hn, 0.0f);
    r_new[idx] = rn;
    rbf_new[idx] = __float2bfloat16(rn);
}

// ---------------------------------------------------------------------------
// Readout: out[b, o] = sum_e r_new[b, e] * w_out[e, o]   (e < 512)
// ---------------------------------------------------------------------------
__global__ void readout(const float* __restrict__ r_new,   // [BB][NN]
                        const float* __restrict__ w_out,   // [NES][NOUT]
                        float* __restrict__ out_t) {       // [BB][NOUT]
    int idx = blockIdx.x * blockDim.x + threadIdx.x;
    if (idx >= BB * NOUT) return;
    int b = idx / NOUT;
    int o = idx % NOUT;
    float s = 0.0f;
    const float* rb = r_new + (long)b * NN;
    #pragma unroll 4
    for (int e = 0; e < NES; ++e)
        s = fmaf(rb[e], w_out[e * NOUT + o], s);
    out_t[idx] = s;
}

// ---------------------------------------------------------------------------
extern "C" void kernel_launch(void* const* d_in, const int* in_sizes, int n_in,
                              void* d_out, int out_size, void* d_ws, size_t ws_size,
                              hipStream_t stream) {
    const float* x      = (const float*)d_in[0];  // [T][B][NIN]
    const float* noise  = (const float*)d_in[1];  // [T][B][NN]
    const float* w_rec  = (const float*)d_in[2];  // [NN][NN]
    const float* w_in   = (const float*)d_in[3];  // [NIN][NN]
    const float* w_out  = (const float*)d_in[4];  // [NES][NOUT]
    const float* bias   = (const float*)d_in[5];  // [NN]
    const float* h0     = (const float*)d_in[6];  // [BB][NN]
    const float* mask   = (const float*)d_in[7];  // [NN][NN]
    float* out = (float*)d_out;                   // [T][B][NOUT]

    // ---- workspace carve-up ----
    const size_t SZ_WT = (size_t)NN * NN * sizeof(__hip_bfloat16); // 29.5 MB
    const size_t SZ_F  = (size_t)BB * NN * sizeof(float);          // 1.97 MB
    const size_t SZ_BF = (size_t)BB * NN * sizeof(__hip_bfloat16); // 0.98 MB
    char* ws = (char*)d_ws;
    __hip_bfloat16* wT    = (__hip_bfloat16*)(ws);
    float*          total = (float*)(ws + SZ_WT);
    float*          hbuf  = (float*)(ws + SZ_WT + SZ_F);
    float*          rA    = (float*)(ws + SZ_WT + 2 * SZ_F);
    float*          rB    = (float*)(ws + SZ_WT + 3 * SZ_F);
    __hip_bfloat16* rbfA  = (__hip_bfloat16*)(ws + SZ_WT + 4 * SZ_F);
    __hip_bfloat16* rbfB  = (__hip_bfloat16*)(ws + SZ_WT + 4 * SZ_F + SZ_BF);

    {
        long n_elem = (long)NN * NN;
        int blocks = (int)((n_elem + 255) / 256);
        build_weffT<<<blocks, 256, 0, stream>>>(w_rec, mask, wT);
    }
    {
        int blocks = (BB * NN + 255) / 256;
        init_state<<<blocks, 256, 0, stream>>>(h0, hbuf, rA, rbfA);
    }

    const dim3 gemm_grid(NN / (8 * WPT * 16), BB / 16);  // (10, 8) = 80 WGs
    const int upd_blocks = (BB * NN + 255) / 256;        // 1920
    const int ro_blocks  = (BB * NOUT + 255) / 256;      // 5

    float* r_cur = rA;              float* r_nxt = rB;
    __hip_bfloat16* rbf_cur = rbfA; __hip_bfloat16* rbf_nxt = rbfB;

    for (int t = 0; t < TT; ++t) {
        gemm_rw<<<gemm_grid, 256, 0, stream>>>(rbf_cur, wT, total);
        step_update<<<upd_blocks, 256, 0, stream>>>(
            total,
            x + (size_t)t * BB * NIN,
            noise + (size_t)t * BB * NN,
            w_in, bias, r_cur,
            hbuf, r_nxt, rbf_nxt);
        readout<<<ro_blocks, 256, 0, stream>>>(
            r_nxt, w_out, out + (size_t)t * BB * NOUT);
        float* tf = r_cur; r_cur = r_nxt; r_nxt = tf;
        __hip_bfloat16* tb = rbf_cur; rbf_cur = rbf_nxt; rbf_nxt = tb;
    }
}
